// TSRNN_81561428951130
// MI455X (gfx1250) — matrix operand order repaired and checked by reading, not compiled, not run on hardware
//
#include <hip/hip_runtime.h>

// ============================================================================
// MI455X (gfx1250) implementation of the dilated-GRU seq2seq reference.
// Strategy: hoist all h-independent matmuls into large parallel bf16 WMMA
// GEMMs (v_wmma_f32_16x16x32_bf16); keep the sequential GRU step minimal with
// weights pre-packed in WMMA B-tile lane layout and resident in VGPRs.
// Round 2: 4 independent WMMA chains in the big GEMMs (2 N-tiles x even/odd K
// partial sums) to hide the bf16 WMMA RAW hazard + B-load latency; prefetch
// next-step gate-input tiles inside the sequential GRU chains.
// ============================================================================

typedef __bf16 bf16_t;
typedef bf16_t v16bf __attribute__((ext_vector_type(16)));
typedef float  v8f   __attribute__((ext_vector_type(8)));
typedef unsigned int u32x4 __attribute__((ext_vector_type(4)));

union Frag { v16bf v; u32x4 q[2]; unsigned short s[16]; };

__device__ inline unsigned short f2bf(float f){
  union { float f; unsigned u; } v; v.f = f;
  unsigned u = v.u + 0x7fffu + ((v.u >> 16) & 1u);
  return (unsigned short)(u >> 16);
}
__device__ inline float bf2f(unsigned short s){
  union { unsigned u; float f; } v; v.u = ((unsigned)s) << 16; return v.f;
}
__device__ inline float sigm(float x){ return 1.f/(1.f + __expf(-x)); }
__device__ inline float tanh_(float x){ return 2.f/(1.f + __expf(-2.f*x)) - 1.f; }
__device__ inline float lrelu(float x){ return x > 0.f ? x : 0.01f*x; }
__device__ inline v8f vzero(){
  v8f z;
#pragma unroll
  for (int i=0;i<8;i++) z[i] = 0.f;
  return z;
}
__device__ inline v8f wmma_bf16(v16bf a, v16bf b, v8f c){
  return __builtin_amdgcn_wmma_f32_16x16x32_bf16(false, a, false, b, (short)0, c, false, false);
}

// ---------------- workspace layout (bytes) ----------------
#define OFF_X0F   0ull                          // [64][512][256] bf16
#define SZ_X0F    16777216ull
#define OFF_X1F   (OFF_X0F + SZ_X0F)            // [64][64][128] bf16
#define SZ_X1F    1048576ull
#define OFF_PK    (OFF_X1F + SZ_X1F)            // packed weights, 3 dils
#define SZ_PK_DIL 589824ull                     // encWiT(196608) encWhT decWiAT decWiBT decWhT (4x98304)
#define OFF_GIE   (OFF_PK + 3*SZ_PK_DIL)        // encoder gate inputs, bf16 D-tiles
#define SZ_GIE_DIL 25165824ull                  // 2048 Mtile-steps * 24 tiles * 512B
#define OFF_GID   (OFF_GIE + 3*SZ_GIE_DIL)      // decoder x1 gate inputs
#define SZ_GID_DIL 3342336ull                   // 272 * 24 * 512
#define OFF_HENC  (OFF_GID + 3*SZ_GID_DIL)      // h_enc fp32 [dil][rows][128]
#define SZ_HENC_DIL 131072ull
#define OFF_DEC   (OFF_HENC + 3*SZ_HENC_DIL)    // dec buffer fp32 [64][64][384]

// ============================================================================
// k_feat0: x0 path (X0 + 2 embeddings -> 40ch -> 128 -> 128) and y0 path
// (1 -> 128 -> 128), written as bf16 [b][t][256] (ex0 ch 0-127, ey ch 128-255).
// ============================================================================
__global__ __launch_bounds__(128) void k_feat0(
    const float* __restrict__ X0, const float* __restrict__ y0,
    const int* __restrict__ tgc0,
    const float* __restrict__ emb0, const float* __restrict__ emb1,
    const float* __restrict__ w1, const float* __restrict__ b1,
    const float* __restrict__ w2, const float* __restrict__ b2,
    const float* __restrict__ yw1, const float* __restrict__ yb1,
    const float* __restrict__ yw2, const float* __restrict__ yb2,
    unsigned short* __restrict__ x0f)
{
  __shared__ float sIn[40];
  __shared__ float sH[128];
  __shared__ float sHy[128];
  int b = blockIdx.x >> 9;
  int t = blockIdx.x & 511;
  int o = threadIdx.x;
  if (o < 8)        sIn[o] = X0[((b<<3) + o)*512 + t];
  else if (o < 24){ int id = tgc0[((b<<9)+t)*2 + 0]; sIn[o] = emb0[id*16 + (o-8)]; }
  else if (o < 40){ int id = tgc0[((b<<9)+t)*2 + 1]; sIn[o] = emb1[id*16 + (o-24)]; }
  float yv = y0[(b<<9) + t];
  sHy[o] = lrelu(yw1[o]*yv + yb1[o]);
  __syncthreads();
  float a1 = b1[o];
#pragma unroll
  for (int c=0;c<40;c++) a1 += w1[o*40 + c]*sIn[c];
  sH[o] = lrelu(a1);
  __syncthreads();
  float a2 = b2[o];
  float a3 = yb2[o];
  for (int c=0;c<128;c++){ a2 += w2[o*128+c]*sH[c]; a3 += yw2[o*128+c]*sHy[c]; }
  size_t base = ((size_t)((b<<9)+t))*256;
  x0f[base + o]       = f2bf(a2);
  x0f[base + 128 + o] = f2bf(a3);
}

// k_feat1: x1 path -> bf16 [b][t][128]
__global__ __launch_bounds__(128) void k_feat1(
    const float* __restrict__ X1, const int* __restrict__ tgc1,
    const float* __restrict__ emb0, const float* __restrict__ emb1,
    const float* __restrict__ w1, const float* __restrict__ b1,
    const float* __restrict__ w2, const float* __restrict__ b2,
    unsigned short* __restrict__ x1f)
{
  __shared__ float sIn[40];
  __shared__ float sH[128];
  int b = blockIdx.x >> 6; int t = blockIdx.x & 63; int o = threadIdx.x;
  if (o < 8)        sIn[o] = X1[((b<<3)+o)*64 + t];
  else if (o < 24){ int id = tgc1[((b<<6)+t)*2+0]; sIn[o] = emb0[id*16 + (o-8)]; }
  else if (o < 40){ int id = tgc1[((b<<6)+t)*2+1]; sIn[o] = emb1[id*16 + (o-24)]; }
  __syncthreads();
  float a1 = b1[o];
#pragma unroll
  for (int c=0;c<40;c++) a1 += w1[o*40+c]*sIn[c];
  sH[o] = lrelu(a1);
  __syncthreads();
  float a2 = b2[o];
  for (int c=0;c<128;c++) a2 += w2[o*128+c]*sH[c];
  x1f[((size_t)((b<<6)+t))*128 + o] = f2bf(a2);
}

// ============================================================================
// k_pack: W (row-major [N][Ksrc] fp32) -> packed bf16 WMMA B tiles (32x16):
// tile (j,k) at (j*NK+k)*1024B; per lane 16 halves; N=lane%16, hi=lane/16;
// halves0-7: K=8*hi+half; halves8-15: K=16+8*hi+(half-8).
// ============================================================================
__global__ __launch_bounds__(256) void k_pack(
    const float* __restrict__ src, char* __restrict__ dst,
    int NK, int Ksrc, int koff, int nItems)
{
  int id = blockIdx.x*256 + threadIdx.x;
  if (id >= nItems) return;
  int lane = id & 31;
  int tile = id >> 5;
  int j = tile / NK;
  int c = tile - j*NK;
  int row = j*16 + (lane & 15);
  int kb  = koff + c*32 + (lane >> 4)*8;
  union { u32x4 q[2]; unsigned short s[16]; } o;
#pragma unroll
  for (int i=0;i<16;i++) o.s[i] = f2bf(src[(size_t)row*Ksrc + kb + i + ((i >> 3) << 3)]);
  u32x4* dp = (u32x4*)(dst + (size_t)tile*1024 + lane*32);
  dp[0] = o.q[0]; dp[1] = o.q[1];
}

// ============================================================================
// k_gi_enc: GI = x_seq @ enc_Wi^T + bi for every (dil, step, 16-row tile).
// A resident (8 chunks); per outer iteration two N-tiles, each split into
// even/odd-K partial accumulators -> 4 independent WMMA chains of length 4.
// Output stored as packed bf16 D tiles (16B/lane/tile).
// ============================================================================
__global__ __launch_bounds__(256) void k_gi_enc(
    const unsigned short* __restrict__ x0f, const char* __restrict__ pk,
    char* __restrict__ gie,
    const float* __restrict__ bi0, const float* __restrict__ bi1,
    const float* __restrict__ bi2)
{
  int task = blockIdx.x*8 + (threadIdx.x>>5);       // 6144 wave tasks
  int lane = threadIdx.x & 31;
  int r   = task >> 11;                             // 2048 per dilation
  int rem = task & 2047;
  int d = 1 << r;
  int Mtiles = 4 << r;
  int t  = rem >> (2 + r);
  int mt = rem & (Mtiles - 1);
  int m = lane & 15, hi = lane >> 4;
  const float* bi = (r==0)?bi0:((r==1)?bi1:bi2);
  int rr = mt*16 + m;
  int p = rr >> 6, bb = rr & 63;
  int ti = t*d + p;
  const unsigned short* arow = x0f + ((size_t)(bb*512 + ti))*256;
  Frag A[8];
#pragma unroll
  for (int c=0;c<8;c++){
    int k0 = c*32 + hi*8;                           // halves0-7 <- K k0..k0+7
    A[c].q[0] = *(const u32x4*)(arow + k0);
    A[c].q[1] = *(const u32x4*)(arow + k0 + 16);    // halves8-15 <- K k0+16..
  }
  const char* wp = pk + (size_t)r*SZ_PK_DIL;        // encWiT
  char* gout = gie + (size_t)r*SZ_GIE_DIL + ((size_t)(t*Mtiles + mt))*24*512;
  for (int j=0;j<24;j+=2){
    v8f D0a = vzero(), D0b = vzero(), D1a = vzero(), D1b = vzero();
#pragma unroll
    for (int c=0;c<8;c+=2){
      Frag B0a, B0b, B1a, B1b;
      const u32x4* p0a = (const u32x4*)(wp + ((size_t)( j     *8 + c    ))*1024 + lane*32);
      const u32x4* p0b = (const u32x4*)(wp + ((size_t)( j     *8 + c + 1))*1024 + lane*32);
      const u32x4* p1a = (const u32x4*)(wp + ((size_t)((j + 1)*8 + c    ))*1024 + lane*32);
      const u32x4* p1b = (const u32x4*)(wp + ((size_t)((j + 1)*8 + c + 1))*1024 + lane*32);
      B0a.q[0] = p0a[0]; B0a.q[1] = p0a[1];
      B0b.q[0] = p0b[0]; B0b.q[1] = p0b[1];
      B1a.q[0] = p1a[0]; B1a.q[1] = p1a[1];
      B1b.q[0] = p1b[0]; B1b.q[1] = p1b[1];
      D0a = wmma_bf16(A[c].v,   B0a.v, D0a);        // 4 independent chains
      D1a = wmma_bf16(A[c].v,   B1a.v, D1a);
      D0b = wmma_bf16(A[c+1].v, B0b.v, D0b);
      D1b = wmma_bf16(A[c+1].v, B1b.v, D1b);
    }
    float bias0 = bi[ j     *16 + m];               // bias per output column N
    float bias1 = bi[(j + 1)*16 + m];
    union { u32x4 q; unsigned short s[8]; } o0, o1;
#pragma unroll
    for (int i=0;i<8;i++){
      o0.s[i] = f2bf(D0a[i] + D0b[i] + bias0);
      o1.s[i] = f2bf(D1a[i] + D1b[i] + bias1);
    }
    *(u32x4*)(gout + (size_t)( j    )*512 + lane*16) = o0.q;
    *(u32x4*)(gout + (size_t)(j + 1)*512 + lane*16) = o1.q;
  }
}

// k_gi_dec: GI = x1_seq @ dec_WiB^T + dec_bi (zero rows past padded length).
// Two N-tiles per iteration -> 2 independent WMMA chains of length 4.
__global__ __launch_bounds__(256) void k_gi_dec(
    const unsigned short* __restrict__ x1f, const char* __restrict__ pk,
    char* __restrict__ gid,
    const float* __restrict__ bi0, const float* __restrict__ bi1,
    const float* __restrict__ bi2)
{
  int task = blockIdx.x*8 + (threadIdx.x>>5);
  if (task >= 796) return;                          // uniform per wave
  int lane = threadIdx.x & 31;
  int r, rem;
  if (task < 260)      { r=0; rem=task; }
  else if (task < 524) { r=1; rem=task-260; }
  else                 { r=2; rem=task-524; }
  int d = 1 << r;
  int Mtiles = 4 << r;
  int t  = rem >> (2 + r);
  int mt = rem & (Mtiles - 1);
  int m = lane & 15, hi = lane >> 4;
  const float* bi = (r==0)?bi0:((r==1)?bi1:bi2);
  int rr = mt*16 + m, p = rr >> 6, bb = rr & 63;
  int ti = t*d + p;                                 // uniform per wave
  Frag A[4];
  if (ti < 64){
    const unsigned short* arow = x1f + ((size_t)(bb*64 + ti))*128;
#pragma unroll
    for (int c=0;c<4;c++){
      int k0 = c*32 + hi*8;
      A[c].q[0] = *(const u32x4*)(arow + k0);
      A[c].q[1] = *(const u32x4*)(arow + k0 + 16);
    }
  } else {
#pragma unroll
    for (int c=0;c<4;c++){
#pragma unroll
      for (int i=0;i<16;i++) A[c].s[i] = 0;
    }
  }
  const char* wp = pk + (size_t)r*SZ_PK_DIL + 393216;   // decWiBT
  char* gout = gid + (size_t)r*SZ_GID_DIL + ((size_t)(t*Mtiles + mt))*24*512;
  for (int j=0;j<24;j+=2){
    v8f D0 = vzero(), D1 = vzero();
#pragma unroll
    for (int c=0;c<4;c++){
      Frag B0, B1;
      const u32x4* p0 = (const u32x4*)(wp + ((size_t)( j     *4 + c))*1024 + lane*32);
      const u32x4* p1 = (const u32x4*)(wp + ((size_t)((j + 1)*4 + c))*1024 + lane*32);
      B0.q[0] = p0[0]; B0.q[1] = p0[1];
      B1.q[0] = p1[0]; B1.q[1] = p1[1];
      D0 = wmma_bf16(A[c].v, B0.v, D0);
      D1 = wmma_bf16(A[c].v, B1.v, D1);
    }
    float bias0 = bi[ j     *16 + m];
    float bias1 = bi[(j + 1)*16 + m];
    union { u32x4 q; unsigned short s[8]; } o0, o1;
#pragma unroll
    for (int i=0;i<8;i++){
      o0.s[i] = f2bf(D0[i] + bias0);
      o1.s[i] = f2bf(D1[i] + bias1);
    }
    *(u32x4*)(gout + (size_t)( j    )*512 + lane*16) = o0.q;
    *(u32x4*)(gout + (size_t)(j + 1)*512 + lane*16) = o1.q;
  }
}

// ============================================================================
// k_rnn_enc: sequential GRU encoder. 28 blocks (one per (dil, 16-row tile)),
// 8 waves; wave w owns units 16w..16w+15 (gate tiles w, 8+w, 16+w).
// Wh B-tiles resident in VGPRs; h kept in regs (D layout), bf16 copy in LDS
// re-read each step in A layout. Next-step GI tiles prefetched
// (global_prefetch) to shorten the serial chain.
// ============================================================================
__global__ __launch_bounds__(256) void k_rnn_enc(
    const char* __restrict__ pk, const char* __restrict__ gie,
    float* __restrict__ henc,
    const float* __restrict__ bh0, const float* __restrict__ bh1,
    const float* __restrict__ bh2)
{
  __shared__ unsigned short hbuf[16*136];           // 16 rows x 128 (+8 pad)
  int bid = blockIdx.x;
  int r, mt;
  if (bid < 4)      { r=0; mt=bid;    }
  else if (bid < 12){ r=1; mt=bid-4;  }
  else              { r=2; mt=bid-12; }
  int Mtiles = 4 << r;
  int L = 512 >> r;
  int w = threadIdx.x >> 5, lane = threadIdx.x & 31;
  int m = lane & 15, hi = lane >> 4;
  const float* bh = (r==0)?bh0:((r==1)?bh1:bh2);
  const char* whp = pk + (size_t)r*SZ_PK_DIL + 196608;  // encWhT
  Frag Bt[3][4];
#pragma unroll
  for (int g=0; g<3; g++){
    int j = g*8 + w;
#pragma unroll
    for (int c=0;c<4;c++){
      const u32x4* bp = (const u32x4*)(whp + ((size_t)(j*4 + c))*1024 + lane*32);
      Bt[g][c].q[0] = bp[0]; Bt[g][c].q[1] = bp[1];
    }
  }
  int u = w*16 + m;
  float bhr = bh[u], bhz = bh[128 + u], bhn = bh[256 + u];
  float h[8];
#pragma unroll
  for (int i=0;i<8;i++) h[i] = 0.f;
  for (int i=threadIdx.x; i<1088; i+=256) ((unsigned*)hbuf)[i] = 0u;
  __syncthreads();
  const char* gbase = gie + (size_t)r*SZ_GIE_DIL;
  for (int t=0;t<L;t++){
    Frag A[4];
#pragma unroll
    for (int c=0;c<4;c++){
      int k0 = c*32 + hi*8;
      A[c].q[0] = *(const u32x4*)(hbuf + m*136 + k0);
      A[c].q[1] = *(const u32x4*)(hbuf + m*136 + k0 + 16);
    }
    const char* gt = gbase + ((size_t)(t*Mtiles + mt))*24*512;
    union { u32x4 q; unsigned short s[8]; } gr, gz, gn;
    gr.q = *(const u32x4*)(gt + (size_t)(w     )*512 + lane*16);
    gz.q = *(const u32x4*)(gt + (size_t)(8  + w)*512 + lane*16);
    gn.q = *(const u32x4*)(gt + (size_t)(16 + w)*512 + lane*16);
    if (t + 1 < L){                                 // prefetch next-step GI
      const char* gn2 = gbase + ((size_t)((t+1)*Mtiles + mt))*24*512;
      __builtin_prefetch(gn2 + (size_t)(w     )*512 + lane*16, 0, 0);
      __builtin_prefetch(gn2 + (size_t)(8  + w)*512 + lane*16, 0, 0);
      __builtin_prefetch(gn2 + (size_t)(16 + w)*512 + lane*16, 0, 0);
    }
    v8f Dr = vzero(), Dz = vzero(), Dn = vzero();
#pragma unroll
    for (int c=0;c<4;c++){                          // 3 independent chains
      Dr = wmma_bf16(A[c].v, Bt[0][c].v, Dr);
      Dz = wmma_bf16(A[c].v, Bt[1][c].v, Dz);
      Dn = wmma_bf16(A[c].v, Bt[2][c].v, Dn);
    }
    __syncthreads();                                 // A reads done
#pragma unroll
    for (int i=0;i<8;i++){
      float rv = sigm(bf2f(gr.s[i]) + Dr[i] + bhr);
      float zv = sigm(bf2f(gz.s[i]) + Dz[i] + bhz);
      float nv = tanh_(bf2f(gn.s[i]) + rv*(Dn[i] + bhn));
      h[i] = (1.f - zv)*nv + zv*h[i];
    }
#pragma unroll
    for (int v=0; v<8; v++){
      int row = v + hi*8;
      hbuf[row*136 + u] = f2bf(h[v]);
    }
    __syncthreads();                                 // writes visible
  }
  float* hout = henc + (size_t)r*32768;
#pragma unroll
  for (int v=0; v<8; v++){
    int row = v + hi*8;
    hout[(size_t)(mt*16 + row)*128 + u] = h[v];
  }
}

// ============================================================================
// k_rnn_dec: sequential GRU decoder (dec_in == h). Per step:
//   gi = h@WiA^T + GIdec (bi included);  gh = h@Wh^T + bh
// Wh resident in VGPRs, WiA tiles streamed (WGP$-resident after step 0).
// Next-step GI tiles prefetched. Results scattered de-dilated into dec[b][t][384].
// ============================================================================
__global__ __launch_bounds__(256) void k_rnn_dec(
    const char* __restrict__ pk, const char* __restrict__ gid,
    const float* __restrict__ henc, float* __restrict__ decb,
    const float* __restrict__ bh0, const float* __restrict__ bh1,
    const float* __restrict__ bh2)
{
  __shared__ unsigned short hbuf[16*136];
  int bid = blockIdx.x;
  int r, mt;
  if (bid < 4)      { r=0; mt=bid;    }
  else if (bid < 12){ r=1; mt=bid-4;  }
  else              { r=2; mt=bid-12; }
  int d = 1 << r;
  int Mtiles = 4 << r;
  int tgt = (64 >> r) + 1;
  int w = threadIdx.x >> 5, lane = threadIdx.x & 31;
  int m = lane & 15, hi = lane >> 4;
  const float* bh = (r==0)?bh0:((r==1)?bh1:bh2);
  const char* whp = pk + (size_t)r*SZ_PK_DIL + 491520;  // decWhT
  const char* wap = pk + (size_t)r*SZ_PK_DIL + 294912;  // decWiAT
  Frag Bh[3][4];
#pragma unroll
  for (int g=0; g<3; g++){
    int j = g*8 + w;
#pragma unroll
    for (int c=0;c<4;c++){
      const u32x4* bp = (const u32x4*)(whp + ((size_t)(j*4 + c))*1024 + lane*32);
      Bh[g][c].q[0] = bp[0]; Bh[g][c].q[1] = bp[1];
    }
  }
  int u = w*16 + m;
  float bhr = bh[u], bhz = bh[128 + u], bhn = bh[256 + u];
  float h[8];
  const float* hin = henc + (size_t)r*32768;
#pragma unroll
  for (int v=0;v<8;v++){ int row = v + hi*8; h[v] = hin[(size_t)(mt*16 + row)*128 + u]; }
#pragma unroll
  for (int v=0;v<8;v++){ int row = v + hi*8; hbuf[row*136 + u] = f2bf(h[v]); }
  __syncthreads();
  int p   = (mt*16) >> 6;
  int bb0 = (mt*16) & 63;
  const char* gbase = gid + (size_t)r*SZ_GID_DIL;
  for (int tt=0; tt<tgt; tt++){
    Frag A[4];
#pragma unroll
    for (int c=0;c<4;c++){
      int k0 = c*32 + hi*8;
      A[c].q[0] = *(const u32x4*)(hbuf + m*136 + k0);
      A[c].q[1] = *(const u32x4*)(hbuf + m*136 + k0 + 16);
    }
    const char* gt = gbase + ((size_t)(tt*Mtiles + mt))*24*512;
    if (tt + 1 < tgt){                              // prefetch next-step GI
      const char* gn2 = gbase + ((size_t)((tt+1)*Mtiles + mt))*24*512;
      __builtin_prefetch(gn2 + (size_t)(w     )*512 + lane*16, 0, 0);
      __builtin_prefetch(gn2 + (size_t)(8  + w)*512 + lane*16, 0, 0);
      __builtin_prefetch(gn2 + (size_t)(16 + w)*512 + lane*16, 0, 0);
    }
    float rv[8], zv[8], hv[8];
    { // gate r
      v8f Dh = vzero(), Di = vzero();
#pragma unroll
      for (int c=0;c<4;c++){
        Frag Bf;
        const u32x4* bp = (const u32x4*)(wap + ((size_t)((w)*4 + c))*1024 + lane*32);
        Bf.q[0] = bp[0]; Bf.q[1] = bp[1];
        Dh = wmma_bf16(A[c].v, Bh[0][c].v, Dh);
        Di = wmma_bf16(A[c].v, Bf.v, Di);
      }
      union { u32x4 q; unsigned short s[8]; } g;
      g.q = *(const u32x4*)(gt + (size_t)(w)*512 + lane*16);
#pragma unroll
      for (int i=0;i<8;i++) rv[i] = sigm(bf2f(g.s[i]) + Di[i] + Dh[i] + bhr);
    }
    { // gate z
      v8f Dh = vzero(), Di = vzero();
#pragma unroll
      for (int c=0;c<4;c++){
        Frag Bf;
        const u32x4* bp = (const u32x4*)(wap + ((size_t)((8+w)*4 + c))*1024 + lane*32);
        Bf.q[0] = bp[0]; Bf.q[1] = bp[1];
        Dh = wmma_bf16(A[c].v, Bh[1][c].v, Dh);
        Di = wmma_bf16(A[c].v, Bf.v, Di);
      }
      union { u32x4 q; unsigned short s[8]; } g;
      g.q = *(const u32x4*)(gt + (size_t)(8 + w)*512 + lane*16);
#pragma unroll
      for (int i=0;i<8;i++) zv[i] = sigm(bf2f(g.s[i]) + Di[i] + Dh[i] + bhz);
    }
    { // gate n + h update
      v8f Dh = vzero(), Di = vzero();
#pragma unroll
      for (int c=0;c<4;c++){
        Frag Bf;
        const u32x4* bp = (const u32x4*)(wap + ((size_t)((16+w)*4 + c))*1024 + lane*32);
        Bf.q[0] = bp[0]; Bf.q[1] = bp[1];
        Dh = wmma_bf16(A[c].v, Bh[2][c].v, Dh);
        Di = wmma_bf16(A[c].v, Bf.v, Di);
      }
      union { u32x4 q; unsigned short s[8]; } g;
      g.q = *(const u32x4*)(gt + (size_t)(16 + w)*512 + lane*16);
#pragma unroll
      for (int i=0;i<8;i++){
        float nv = tanh_(bf2f(g.s[i]) + Di[i] + rv[i]*(Dh[i] + bhn));
        hv[i] = (1.f - zv[i])*nv + zv[i]*h[i];
      }
    }
#pragma unroll
    for (int i=0;i<8;i++) h[i] = hv[i];
    __syncthreads();
#pragma unroll
    for (int v=0;v<8;v++){ int row = v + hi*8; hbuf[row*136 + u] = f2bf(h[v]); }
    int tout = tt*d + p;                            // uniform per block
    if (tout < 64){
#pragma unroll
      for (int v=0;v<8;v++){
        int row = v + hi*8;
        decb[((size_t)(bb0 + row)*64 + tout)*384 + r*128 + u] = h[v];
      }
    }
    __syncthreads();
  }
}

// ============================================================================
// k_out: final 384 -> 128 (leaky) -> 1 pointwise convs; one block per (b,t).
// ============================================================================
__global__ __launch_bounds__(128) void k_out(
    const float* __restrict__ decb,
    const float* __restrict__ w1, const float* __restrict__ b1,
    const float* __restrict__ w2, const float* __restrict__ b2,
    float* __restrict__ out)
{
  __shared__ float sDec[384];
  __shared__ float sRed[128];
  int b = blockIdx.x >> 6, t = blockIdx.x & 63, o = threadIdx.x;
  const float* drow = decb + ((size_t)(b*64 + t))*384;
  for (int i=o; i<384; i+=128) sDec[i] = drow[i];
  __syncthreads();
  float a = b1[o];
  for (int c=0;c<384;c++) a += w1[o*384 + c]*sDec[c];
  sRed[o] = w2[o]*lrelu(a);
  __syncthreads();
  for (int s=64; s>0; s>>=1){
    if (o < s) sRed[o] += sRed[o+s];
    __syncthreads();
  }
  if (o == 0) out[b*64 + t] = sRed[0] + b2[0];
}

// ============================================================================
// host launch
// ============================================================================
extern "C" void kernel_launch(void* const* d_in, const int* in_sizes, int n_in,
                              void* d_out, int out_size, void* d_ws, size_t ws_size,
                              hipStream_t stream)
{
  (void)in_sizes; (void)n_in; (void)out_size; (void)ws_size;
  auto F = [&](int i){ return (const float*)d_in[i]; };
  auto I = [&](int i){ return (const int*)d_in[i]; };
  char* ws = (char*)d_ws;
  unsigned short* x0f = (unsigned short*)(ws + OFF_X0F);
  unsigned short* x1f = (unsigned short*)(ws + OFF_X1F);
  char*  pk   = ws + OFF_PK;
  char*  gie  = ws + OFF_GIE;
  char*  gid  = ws + OFF_GID;
  float* henc = (float*)(ws + OFF_HENC);
  float* decb = (float*)(ws + OFF_DEC);

  // input index map (setup_inputs dict order, nested in insertion order):
  // 0:X0 1:y0 2:X1 3:tgc0 4:tgc1 5:emb0 6:emb1
  // 7-10: ex0_w1,b1,w2,b2   11-14: ex1_*   15-18: ey_*
  // 19+8r: enc_Wi,enc_Wh,enc_bi,enc_bh,dec_Wi,dec_Wh,dec_bi,dec_bh
  // 43: dc_w1 44: dc_b1 45: dc_w2 46: dc_b2

  k_feat0<<<64*512, 128, 0, stream>>>(F(0), F(1), I(3), F(5), F(6),
      F(7), F(8), F(9), F(10), F(15), F(16), F(17), F(18), x0f);
  k_feat1<<<64*64, 128, 0, stream>>>(F(2), I(4), F(5), F(6),
      F(11), F(12), F(13), F(14), x1f);
  for (int r=0;r<3;r++){
    int pb = 19 + r*8;
    char* base = pk + (size_t)r*SZ_PK_DIL;
    k_pack<<<24, 256, 0, stream>>>(F(pb+0), base + 0,      8, 256, 0,   6144); // encWiT
    k_pack<<<12, 256, 0, stream>>>(F(pb+1), base + 196608, 4, 128, 0,   3072); // encWhT
    k_pack<<<12, 256, 0, stream>>>(F(pb+4), base + 294912, 4, 256, 0,   3072); // decWiAT
    k_pack<<<12, 256, 0, stream>>>(F(pb+4), base + 393216, 4, 256, 128, 3072); // decWiBT
    k_pack<<<12, 256, 0, stream>>>(F(pb+5), base + 491520, 4, 128, 0,   3072); // decWhT
  }
  k_gi_enc<<<768, 256, 0, stream>>>(x0f, pk, gie, F(21), F(29), F(37));
  k_gi_dec<<<100, 256, 0, stream>>>(x1f, pk, gid, F(25), F(33), F(41));
  k_rnn_enc<<<28, 256, 0, stream>>>(pk, gie, henc, F(22), F(30), F(38));
  k_rnn_dec<<<28, 256, 0, stream>>>(pk, gid, henc, decb, F(26), F(34), F(42));
  k_out<<<64*64, 128, 0, stream>>>(decb, F(43), F(44), F(45), F(46), (float*)d_out);
}
